// DigitalTwinSimulator_26860725469303
// MI455X (gfx1250) — matrix operand order repaired and checked by reading, not compiled, not run on hardware
//
#include <hip/hip_runtime.h>

// ---------------- problem constants ----------------
#define BATCH   4096
#define T_ENC   128
#define DIN     4
#define HID     64
#define NSTEPS  60
#define GATES   256      // 4*HID
#define NTILE   16       // GATES/16 gate column tiles
#define WAVES   8
#define BLOCKT  (WAVES * 32)

typedef __attribute__((ext_vector_type(16))) __bf16 v16bf;
typedef __attribute__((ext_vector_type(8)))  float  v8f;

// fragment = 32 lanes x 16 bf16 = 512 elements (1KB)
#define FRAG_ELEMS 512

// workspace / LDS weight-block layout (bytes) -- identical in d_ws and LDS
#define OFF_WHH0   0          // 16 tiles * 2 ksteps * 512 bf16 = 32768 B
#define OFF_WIH1   32768
#define OFF_WHH1   65536
#define OFF_WIH0   98304      // 16 tiles * 1 kstep  * 512 bf16 = 16384 B (bias0 folded at k=4)
#define OFF_WFC    114688     // 2 ksteps * 512 bf16 = 2048 B
#define OFF_BIAS0  116736     // 256 f32 (unused by gates now; kept for layout stability)
#define OFF_BIAS1  117760     // 256 f32
#define OFF_BFC    118784     // 4 f32 (pad to 16)
#define WBLK_BYTES 118800     // multiple of 16

// ---------------- fast activations ----------------
#if __has_builtin(__builtin_amdgcn_tanhf)
__device__ __forceinline__ float tanh_(float x) { return __builtin_amdgcn_tanhf(x); }
#else
__device__ __forceinline__ float tanh_(float x) {
    return 2.0f * __builtin_amdgcn_rcpf(1.0f + __expf(-2.0f * x)) - 1.0f;
}
#endif
__device__ __forceinline__ float sigm(float x) { return fmaf(0.5f, tanh_(0.5f * x), 0.5f); }

__device__ __forceinline__ v16bf loadB(const __bf16* base, int fragIdx, int lane) {
    return *(const v16bf*)(base + fragIdx * FRAG_ELEMS + lane * 16);
}

__device__ __forceinline__ v8f wmma_bf16(v16bf a, v16bf b, v8f c) {
    return __builtin_amdgcn_wmma_f32_16x16x32_bf16(false, a, false, b, (short)0, c, false, false);
}

// ---------------- prep kernel: swizzle weights into B-fragment layout ----------------
// B-matrix 32x16 bf16 layout: lane (0..31), slot (0..15 within v16bf):
//   n = tile*16 + (lane & 15)
//   k = kstep*32 + slot + (lane>=16 ? 16 : 0)
// value = W[n, k]   (gates = in @ W^T  =>  B[k][n] = W[n][k])
__device__ __forceinline__ void fillK64(__bf16* dst, const float* W, int i) {
    int tile = i >> 10;           // /1024  (2 ksteps * 512)
    int r    = i & 1023;
    int ks   = r >> 9;
    int q    = r & 511;
    int ln   = q >> 4;
    int sl   = q & 15;
    int n    = tile * 16 + (ln & 15);
    int k    = ks * 32 + sl + ((ln >= 16) ? 16 : 0);
    dst[i]   = (__bf16)W[n * HID + k];
}

#define NB_K64 (NTILE * 2 * FRAG_ELEMS)   // 16384
#define NB_K4  (NTILE * 1 * FRAG_ELEMS)   // 8192
#define NB_FC  (2 * FRAG_ELEMS)           // 1024
#define PREP_TOTAL (3 * NB_K64 + NB_K4 + NB_FC + GATES + GATES + DIN)

__global__ void prep_kernel(const float* Wih0, const float* Whh0,
                            const float* Wih1, const float* Whh1,
                            const float* Wfc,
                            const float* bih0, const float* bhh0,
                            const float* bih1, const float* bhh1,
                            const float* bfc,
                            __bf16* fWhh0, __bf16* fWih1, __bf16* fWhh1,
                            __bf16* fWih0, __bf16* fWfc,
                            float* bias0, float* bias1, float* bfc_o) {
    int i = blockIdx.x * blockDim.x + threadIdx.x;
    if (i >= PREP_TOTAL) return;

    if (i < NB_K64) { fillK64(fWhh0, Whh0, i); return; }
    i -= NB_K64;
    if (i < NB_K64) { fillK64(fWih1, Wih1, i); return; }
    i -= NB_K64;
    if (i < NB_K64) { fillK64(fWhh1, Whh1, i); return; }
    i -= NB_K64;
    if (i < NB_K4) {          // Wih0: [256,4], K padded to 32; bias0 folded at k=4
        int tile = i >> 9;
        int q    = i & 511;
        int ln   = q >> 4;
        int sl   = q & 15;
        int n    = tile * 16 + (ln & 15);
        int k    = sl + ((ln >= 16) ? 16 : 0);
        float v  = 0.0f;
        if (k < DIN)       v = Wih0[n * DIN + k];
        else if (k == DIN) v = bih0[n] + bhh0[n];   // bias row, paired with ax[4]=1.0
        fWih0[i] = (__bf16)v;
        return;
    }
    i -= NB_K4;
    if (i < NB_FC) {          // Wfc: [4,64], N zero-padded to 16, 2 ksteps
        int ks = i >> 9;
        int q  = i & 511;
        int ln = q >> 4;
        int sl = q & 15;
        int n  = ln & 15;
        int k  = ks * 32 + sl + ((ln >= 16) ? 16 : 0);
        fWfc[i] = (n < DIN) ? (__bf16)Wfc[n * HID + k] : (__bf16)0.0f;
        return;
    }
    i -= NB_FC;
    if (i < GATES) { bias0[i] = bih0[i] + bhh0[i]; return; }
    i -= GATES;
    if (i < GATES) { bias1[i] = bih1[i] + bhh1[i]; return; }
    i -= GATES;
    bfc_o[i] = bfc[i];
}

// ---------------- main LSTM kernel ----------------
__global__ __launch_bounds__(BLOCKT)
void lstm_kernel(const float* __restrict__ xg,
                 const uint4* __restrict__ wsrc,   // swizzled weight block in d_ws
                 float* __restrict__ out) {
    // weights + biases live in LDS (copied once per block)
    __shared__ uint4 wlds4[WBLK_BYTES / 16];
    __shared__ __align__(16) __bf16 hstageAll[WAVES][16][HID];  // per-wave D->A staging
    __shared__ __align__(16) float  xstageAll[WAVES][16][DIN];

    for (int i = threadIdx.x; i < WBLK_BYTES / 16; i += BLOCKT) wlds4[i] = wsrc[i];
    __syncthreads();

    const char* wl = (const char*)wlds4;
    const __bf16* fWhh0 = (const __bf16*)(wl + OFF_WHH0);
    const __bf16* fWih1 = (const __bf16*)(wl + OFF_WIH1);
    const __bf16* fWhh1 = (const __bf16*)(wl + OFF_WHH1);
    const __bf16* fWih0 = (const __bf16*)(wl + OFF_WIH0);
    const __bf16* fWfc  = (const __bf16*)(wl + OFF_WFC);
    const float*  bias1 = (const float*)(wl + OFF_BIAS1);
    const float*  bfc   = (const float*)(wl + OFF_BFC);

    const int lane   = threadIdx.x & 31;
    const int wave   = threadIdx.x >> 5;
    const int m      = lane & 15;                 // A-frag row / D-frag column index
    const int hiOff  = (lane >> 4) ? 8 : 0;       // D-frag row offset, A-frag K offset is 2*hiOff
    const int b0     = (blockIdx.x * WAVES + wave) * 16;

    __bf16* stage = &hstageAll[wave][0][0];
    float*  xst   = &xstageAll[wave][0][0];

    v16bf zb;
#pragma unroll
    for (int j = 0; j < 16; j++) zb[j] = (__bf16)0.0f;
    v8f zf;
#pragma unroll
    for (int r = 0; r < 8; r++) zf[r] = 0.0f;

    // persistent recurrent state (registers across all 188 steps)
    v16bf aH0[2] = {zb, zb};
    v16bf aH1[2] = {zb, zb};
    v8f   c0[4]  = {zf, zf, zf, zf};
    v8f   c1[4]  = {zf, zf, zf, zf};
    v16bf ax     = zb;                            // x A-frag: slots 0-3 = x, slot 4 = 1.0 (bias row)
    if (hiOff == 0) ax[4] = (__bf16)1.0f;

    // One LSTM layer step.  bias==nullptr => bias is folded into the x-side
    // weight fragment and the accumulator chain starts from inline C=0.
    auto layer_step = [&](const v16bf* aX, int nx, const __bf16* wX,
                          v16bf* aH, const __bf16* wH,
                          const float* bias, v8f* c) {
#pragma unroll
        for (int jj = 0; jj < 4; jj++) {
            // keep fragment loads inside this scope (no hoisting / register blowup)
            asm volatile("" ::: "memory");
            v8f acc[4];
#pragma unroll
            for (int g = 0; g < 4; g++) {         // gate order: i, f, g, o
                int tile = jj + g * 4;
                v8f a = {0.f, 0.f, 0.f, 0.f, 0.f, 0.f, 0.f, 0.f};  // -> inline SRC2=0
                if (bias) {
                    float bv = bias[tile * 16 + m];
#pragma unroll
                    for (int r = 0; r < 8; r++) a[r] = bv;
                }
                for (int ks = 0; ks < nx; ks++)
                    a = wmma_bf16(aX[ks], loadB(wX, tile * nx + ks, lane), a);
#pragma unroll
                for (int ks = 0; ks < 2; ks++)
                    a = wmma_bf16(aH[ks], loadB(wH, tile * 2 + ks, lane), a);
                acc[g] = a;
            }
            v8f cn;
            int ncol = jj * 16 + m;
#pragma unroll
            for (int r = 0; r < 8; r++) {
                float iv = sigm(acc[0][r]);
                float fv = sigm(acc[1][r]);
                float gv = tanh_(acc[2][r]);
                float ov = sigm(acc[3][r]);
                float cv = fmaf(fv, c[jj][r], iv * gv);
                cn[r] = cv;
                stage[(r + hiOff) * HID + ncol] = (__bf16)(ov * tanh_(cv));
            }
            c[jj] = cn;
        }
        // rebuild A-fragments of h: each lane reads 16 contiguous bf16 in the
        // stage row (contiguous-K layout, k = 16h + s) -> two ds_load_b128 per fragment
#pragma unroll
        for (int ks = 0; ks < 2; ks++) {
            union { uint4 u[2]; v16bf v; } cvt;
            cvt.u[0] = *(const uint4*)(stage + m * HID + ks * 32 + 2 * hiOff);
            cvt.u[1] = *(const uint4*)(stage + m * HID + ks * 32 + 8 + 2 * hiOff);
            aH[ks] = cvt.v;
        }
    };

    // -------- encode phase: scan over T time steps --------
    for (int t = 0; t < T_ENC; t++) {
        asm volatile("" ::: "memory");           // block LICM across time steps
        if (hiOff == 0) {                        // lanes 0-15 carry row m, K slots 0-3
            float4 xv = *(const float4*)(xg + ((size_t)(b0 + m) * T_ENC + t) * DIN);
            ax[0] = (__bf16)xv.x; ax[1] = (__bf16)xv.y;
            ax[2] = (__bf16)xv.z; ax[3] = (__bf16)xv.w;
        }
        layer_step(&ax, 1, fWih0, aH0, fWhh0, nullptr, c0);   // bias0 folded into fWih0
        layer_step(aH0, 2, fWih1, aH1, fWhh1, bias1, c1);
    }

    // -------- autoregressive rollout (first input = x[:, T-1, :], already in ax) --------
    for (int s = 0; s < NSTEPS; s++) {
        asm volatile("" ::: "memory");
        layer_step(&ax, 1, fWih0, aH0, fWhh0, nullptr, c0);
        layer_step(aH0, 2, fWih1, aH1, fWhh1, bias1, c1);

        // pred[16,4] = h1 @ Wfc^T + bfc   (N padded to 16; cols 0-3 valid)
        float pb = (m < DIN) ? bfc[m] : 0.0f;
        v8f p;
#pragma unroll
        for (int r = 0; r < 8; r++) p[r] = pb;
#pragma unroll
        for (int ks = 0; ks < 2; ks++)
            p = wmma_bf16(aH1[ks], loadB(fWfc, ks, lane), p);

        if (m < DIN) {
#pragma unroll
            for (int r = 0; r < 8; r++) {
                int row = r + hiOff;
                out[(size_t)(b0 + row) * (NSTEPS * DIN) + (size_t)s * DIN + m] = p[r];
                xst[row * DIN + m] = p[r];       // feed back as next input
            }
        }
        if (hiOff == 0) {
            float4 xv = *(const float4*)(xst + m * DIN);
            ax[0] = (__bf16)xv.x; ax[1] = (__bf16)xv.y;
            ax[2] = (__bf16)xv.z; ax[3] = (__bf16)xv.w;
        }
    }
}

extern "C" void kernel_launch(void* const* d_in, const int* in_sizes, int n_in,
                              void* d_out, int out_size, void* d_ws, size_t ws_size,
                              hipStream_t stream) {
    const float* x    = (const float*)d_in[0];
    const float* Wih0 = (const float*)d_in[1];
    const float* Whh0 = (const float*)d_in[2];
    const float* bih0 = (const float*)d_in[3];
    const float* bhh0 = (const float*)d_in[4];
    const float* Wih1 = (const float*)d_in[5];
    const float* Whh1 = (const float*)d_in[6];
    const float* bih1 = (const float*)d_in[7];
    const float* bhh1 = (const float*)d_in[8];
    const float* Wfc  = (const float*)d_in[9];
    const float* bfc  = (const float*)d_in[10];

    char* ws = (char*)d_ws;
    __bf16* fWhh0 = (__bf16*)(ws + OFF_WHH0);
    __bf16* fWih1 = (__bf16*)(ws + OFF_WIH1);
    __bf16* fWhh1 = (__bf16*)(ws + OFF_WHH1);
    __bf16* fWih0 = (__bf16*)(ws + OFF_WIH0);
    __bf16* fWfc  = (__bf16*)(ws + OFF_WFC);
    float*  bias0 = (float*)(ws + OFF_BIAS0);
    float*  bias1 = (float*)(ws + OFF_BIAS1);
    float*  bfc_o = (float*)(ws + OFF_BFC);

    prep_kernel<<<(PREP_TOTAL + 255) / 256, 256, 0, stream>>>(
        Wih0, Whh0, Wih1, Whh1, Wfc, bih0, bhh0, bih1, bhh1, bfc,
        fWhh0, fWih1, fWhh1, fWih0, fWfc, bias0, bias1, bfc_o);

    lstm_kernel<<<BATCH / (16 * WAVES), BLOCKT, 0, stream>>>(
        x, (const uint4*)d_ws, (float*)d_out);
}
